// GCN_45646912422072
// MI455X (gfx1250) — matrix operand order repaired and checked by reading, not compiled, not run on hardware
//
#include <hip/hip_runtime.h>
#include <hip/hip_bf16.h>

// ---------------------------------------------------------------------------
// Types for CDNA5 WMMA (wave32)
// ---------------------------------------------------------------------------
typedef __bf16 bf16;
typedef __attribute__((ext_vector_type(16))) __bf16 v16bf;
typedef __attribute__((ext_vector_type(8)))  __bf16 bf16x8;
typedef __attribute__((ext_vector_type(4)))  __bf16 bf16x4;
typedef __attribute__((ext_vector_type(8)))  float  v8f;
typedef __attribute__((ext_vector_type(4)))  float  v4f;

union Frag {
    bf16x8 h[2];
    v16bf  v;
};

#define GCN_N 100000
#define GCN_E 800000
#define GCN_H 128

#define WMMA_BF16(A, B, C) \
    __builtin_amdgcn_wmma_f32_16x16x32_bf16(false, (A), false, (B), (short)0, (C), false, false)

// ---------------------------------------------------------------------------
// fp32 -> bf16 conversion, 4 elements / thread (n must be a multiple of 4)
// ---------------------------------------------------------------------------
__global__ void f2bf_kernel(const float* __restrict__ in, bf16* __restrict__ out, int n4)
{
    int t = blockIdx.x * blockDim.x + threadIdx.x;
    if (t < n4) {
        v4f v = ((const v4f*)in)[t];
        bf16x4 o;
        o[0] = (bf16)v[0]; o[1] = (bf16)v[1]; o[2] = (bf16)v[2]; o[3] = (bf16)v[3];
        ((bf16x4*)out)[t] = o;
    }
}

// ---------------------------------------------------------------------------
// WMMA GEMM:  Y[m][n] = sum_k A[m][k] * W[n][k] (+ A2@W2) + bias[n]
//   A:  [nrows x 128] bf16 row-major
//   W:  [OUTW x 128]  bf16 row-major (i.e. we compute A @ W^T)
// 8 waves / block, each wave computes a 16-row strip over NT 16-col tiles.
// K = 128 = 4 x (16x16x32 bf16 WMMA), run as two independent acc chains.
// ---------------------------------------------------------------------------
template<int NT, bool HAS2, bool DO_RELU, bool STATS, bool BF16OUT>
__global__ __launch_bounds__(256) void gemm_wmma(
    const bf16* __restrict__ A,  const bf16* __restrict__ W,
    const bf16* __restrict__ A2, const bf16* __restrict__ W2,
    const float* __restrict__ bias,
    float* __restrict__ outF, bf16* __restrict__ outB,
    float* __restrict__ colsum, float* __restrict__ colssq,
    int nrows)
{
    constexpr int OUTW = NT * 16;

    __shared__ float s_sum[128];
    __shared__ float s_ssq[128];
    if (STATS) {
        for (int t = threadIdx.x; t < 128; t += 256) { s_sum[t] = 0.f; s_ssq[t] = 0.f; }
        __syncthreads();
    }

    const int wave = threadIdx.x >> 5;   // wave32
    const int lane = threadIdx.x & 31;
    const int half = lane >> 4;          // 0: lanes 0-15, 1: lanes 16-31
    const int l16  = lane & 15;

    const int rowBase = blockIdx.x * 128 + wave * 16;
    // wave-uniform: does this wave's 16-row strip lie fully inside the matrix?
    const bool fullTile = (rowBase + 16 <= nrows);

    const int m  = rowBase + l16;        // A-matrix row for this lane (both halves)
    const bool mv = (m < nrows);
    const size_t mc = (size_t)(mv ? m : (nrows - 1));

    // --- load A (and A2) fragments in contiguous k order (same as B) -----------
    // lane(half,m): vgpr0..3 hold K = kk + half*16 + {0..7}
    //               vgpr4..7 hold K = kk + half*16 + 8 + {0..7}
    Frag a[4];
    Frag a2[4];
    if (mv) {
        const bf16* ar = A + mc * GCN_H;
        #pragma unroll
        for (int kk4 = 0; kk4 < 4; ++kk4) {
            const int kk = kk4 * 32;
            a[kk4].h[0] = *(const bf16x8*)(ar + kk + half * 16);
            a[kk4].h[1] = *(const bf16x8*)(ar + kk + 8 + half * 16);
        }
        if (HAS2) {
            const bf16* ar2 = A2 + mc * GCN_H;
            #pragma unroll
            for (int kk4 = 0; kk4 < 4; ++kk4) {
                const int kk = kk4 * 32;
                a2[kk4].h[0] = *(const bf16x8*)(ar2 + kk + half * 16);
                a2[kk4].h[1] = *(const bf16x8*)(ar2 + kk + 8 + half * 16);
            }
        }
    } else {
        #pragma unroll
        for (int kk4 = 0; kk4 < 4; ++kk4) {
            #pragma unroll
            for (int i = 0; i < 16; ++i) a[kk4].v[i] = (bf16)0.0f;
            if (HAS2) {
                #pragma unroll
                for (int i = 0; i < 16; ++i) a2[kk4].v[i] = (bf16)0.0f;
            }
        }
    }

    #pragma unroll
    for (int nt = 0; nt < NT; ++nt) {
        const int n = nt * 16 + l16;     // output column for this lane
        const float bcol = bias[n];

        // --- preload all B fragments for this tile (one load clause) -------
        // B in contiguous k order (same as A): lanes 0-15 hold K=0..15, lanes 16-31 K=16..31
        // b[j] = W[n][kk + half*16 + j]  -> 32 contiguous bytes of the weight row.
        const bf16* wr = W + (size_t)n * GCN_H + half * 16;
        v16bf b0 = *(const v16bf*)(wr);
        v16bf b1 = *(const v16bf*)(wr + 32);
        v16bf b2 = *(const v16bf*)(wr + 64);
        v16bf b3 = *(const v16bf*)(wr + 96);

        // two independent accumulator chains to break the RAW WMMA chain
        v8f acc0, acc1;
        #pragma unroll
        for (int i = 0; i < 8; ++i) { acc0[i] = 0.f; acc1[i] = 0.f; }

        acc0 = WMMA_BF16(a[0].v, b0, acc0);
        acc1 = WMMA_BF16(a[1].v, b1, acc1);
        acc0 = WMMA_BF16(a[2].v, b2, acc0);
        acc1 = WMMA_BF16(a[3].v, b3, acc1);

        if (HAS2) {
            const bf16* wr2 = W2 + (size_t)n * GCN_H + half * 16;
            v16bf c0 = *(const v16bf*)(wr2);
            v16bf c1 = *(const v16bf*)(wr2 + 32);
            v16bf c2 = *(const v16bf*)(wr2 + 64);
            v16bf c3 = *(const v16bf*)(wr2 + 96);
            acc0 = WMMA_BF16(a2[0].v, c0, acc0);
            acc1 = WMMA_BF16(a2[1].v, c1, acc1);
            acc0 = WMMA_BF16(a2[2].v, c2, acc0);
            acc1 = WMMA_BF16(a2[3].v, c3, acc1);
        }

        float s = 0.f, q = 0.f;
        // C/D layout: vgpr i -> row M = half*8 + i, col N = l16
        const int r0 = rowBase + half * 8;
        if (fullTile) {
            // branch-free epilogue, stores via one base ptr + immediate offsets
            if (BF16OUT) {
                bf16* op = outB + (size_t)r0 * OUTW + n;
                #pragma unroll
                for (int i = 0; i < 8; ++i) {
                    float v = acc0[i] + acc1[i] + bcol;
                    if (DO_RELU) v = v > 0.f ? v : 0.f;
                    op[(size_t)i * OUTW] = (bf16)v;
                    if (STATS) { s += v; q += v * v; }
                }
            } else {
                float* op = outF + (size_t)r0 * OUTW + n;
                #pragma unroll
                for (int i = 0; i < 8; ++i) {
                    float v = acc0[i] + acc1[i] + bcol;
                    if (DO_RELU) v = v > 0.f ? v : 0.f;
                    op[(size_t)i * OUTW] = v;
                    if (STATS) { s += v; q += v * v; }
                }
            }
        } else {
            #pragma unroll
            for (int i = 0; i < 8; ++i) {
                const int r = r0 + i;
                float v = acc0[i] + acc1[i] + bcol;
                if (DO_RELU) v = v > 0.f ? v : 0.f;
                if (r < nrows) {
                    if (BF16OUT) outB[(size_t)r * OUTW + n] = (bf16)v;
                    else         outF[(size_t)r * OUTW + n] = v;
                    if (STATS) { s += v; q += v * v; }
                }
            }
        }
        if (STATS) {
            atomicAdd(&s_sum[n], s);     // ds_add_f32
            atomicAdd(&s_ssq[n], q);
        }
    }

    if (STATS) {
        __syncthreads();
        for (int t = threadIdx.x; t < 128; t += 256) {
            unsafeAtomicAdd(&colsum[t], s_sum[t]);   // global_atomic_add_f32
            unsafeAtomicAdd(&colssq[t], s_ssq[t]);
        }
    }
}

// ---------------------------------------------------------------------------
// BatchNorm finalize: per-column scale/shift from accumulated sums
// ---------------------------------------------------------------------------
__global__ void bn_finalize_kernel(const float* __restrict__ colsum,
                                   const float* __restrict__ colssq,
                                   const float* __restrict__ gamma,
                                   const float* __restrict__ beta,
                                   float* __restrict__ scale,
                                   float* __restrict__ shift, int nrows)
{
    int c = threadIdx.x;
    if (c < 128) {
        float invN = 1.0f / (float)nrows;
        float m = colsum[c] * invN;
        float v = colssq[c] * invN - m * m;
        float sc = gamma[c] * rsqrtf(v + 1e-5f);
        scale[c] = sc;
        shift[c] = beta[c] - m * sc;
    }
}

// BN apply + relu, emit bf16 activations; 4 elements / thread
__global__ void bn_apply_relu_kernel(const float* __restrict__ Y,
                                     const float* __restrict__ scale,
                                     const float* __restrict__ shift,
                                     bf16* __restrict__ X, long total4)
{
    long t = (long)blockIdx.x * blockDim.x + threadIdx.x;
    if (t < total4) {
        int c = (int)((t * 4) & 127);            // 4 consecutive columns
        v4f y = ((const v4f*)Y)[t];
        bf16x4 o;
        #pragma unroll
        for (int i = 0; i < 4; ++i) {
            float v = y[i] * scale[c + i] + shift[c + i];
            v = v > 0.f ? v : 0.f;
            o[i] = (bf16)v;
        }
        ((bf16x4*)X)[t] = o;
    }
}

// ---------------------------------------------------------------------------
// Graph aggregation
// ---------------------------------------------------------------------------
__global__ void deg_kernel(const int* __restrict__ dst, float* __restrict__ deg, int E)
{
    int t = blockIdx.x * blockDim.x + threadIdx.x;
    if (t < E) unsafeAtomicAdd(&deg[dst[t]], 1.0f);
}

// One wave per edge; each lane moves 4 features (128 total) with f32 atomics.
__global__ __launch_bounds__(256) void agg_kernel(
    const bf16* __restrict__ X, const int* __restrict__ src,
    const int* __restrict__ dst, float* __restrict__ AGG, int E)
{
    int e = blockIdx.x * 8 + (threadIdx.x >> 5);
    if (e >= E) return;
    int lane = threadIdx.x & 31;
    int s = src[e], d = dst[e];
    bf16x4 v = *(const bf16x4*)(X + (size_t)s * GCN_H + lane * 4);
    float* o = AGG + (size_t)d * GCN_H + lane * 4;
    unsafeAtomicAdd(o + 0, (float)v[0]);
    unsafeAtomicAdd(o + 1, (float)v[1]);
    unsafeAtomicAdd(o + 2, (float)v[2]);
    unsafeAtomicAdd(o + 3, (float)v[3]);
}

// mean = agg / max(deg, 1), emitted as bf16; 4 elements / thread
__global__ void mean_kernel(const float* __restrict__ AGG, const float* __restrict__ deg,
                            bf16* __restrict__ M, long total4)
{
    long t = (long)blockIdx.x * blockDim.x + threadIdx.x;
    if (t < total4) {
        int r = (int)((t * 4) >> 7);             // same row for all 4 elements
        float d = deg[r];
        d = d > 1.f ? d : 1.f;
        float inv = 1.0f / d;
        v4f v = ((const v4f*)AGG)[t];
        bf16x4 o;
        #pragma unroll
        for (int i = 0; i < 4; ++i) o[i] = (bf16)(v[i] * inv);
        ((bf16x4*)M)[t] = o;
    }
}

// ---------------------------------------------------------------------------
// Launcher
// ---------------------------------------------------------------------------
extern "C" void kernel_launch(void* const* d_in, const int* in_sizes, int n_in,
                              void* d_out, int out_size, void* d_ws, size_t ws_size,
                              hipStream_t stream)
{
    const int N = GCN_N, E = GCN_E, H = GCN_H;
    const long NH = (long)N * H;

    // inputs (setup_inputs order)
    const float* h   = (const float*)d_in[0];
    const float* W1  = (const float*)d_in[1];
    const float* b1  = (const float*)d_in[2];
    const float* g1  = (const float*)d_in[3];
    const float* bt1 = (const float*)d_in[4];
    const float* Ws1 = (const float*)d_in[5];
    const float* Wn1 = (const float*)d_in[6];
    const float* bs1 = (const float*)d_in[7];
    const float* W2  = (const float*)d_in[8];
    const float* b2  = (const float*)d_in[9];
    const float* g2  = (const float*)d_in[10];
    const float* bt2 = (const float*)d_in[11];
    const float* Ws2 = (const float*)d_in[12];
    const float* Wn2 = (const float*)d_in[13];
    const float* bs2 = (const float*)d_in[14];
    const float* W3  = (const float*)d_in[15];
    const float* b3  = (const float*)d_in[16];
    const int*   src = (const int*)d_in[17];
    const int*   dst = (const int*)d_in[18];

    // workspace carve-up (256B aligned)
    char* wsp = (char*)d_ws;
    auto alloc = [&](size_t bytes) -> char* {
        char* p = wsp;
        wsp += (bytes + 255) & ~(size_t)255;
        return p;
    };
    float* big    = (float*)alloc(NH * sizeof(float));   // pre-BN Y / AGG (ping-pong)
    bf16*  bfA    = (bf16*) alloc(NH * sizeof(bf16));    // activation ping
    bf16*  bfB    = (bf16*) alloc(NH * sizeof(bf16));    // activation pong
    bf16*  bfM    = (bf16*) alloc(NH * sizeof(bf16));    // neighbor mean
    float* degv   = (float*)alloc((size_t)N * sizeof(float));
    float* colsum = (float*)alloc(128 * sizeof(float));
    float* colssq = (float*)alloc(128 * sizeof(float));
    float* scale  = (float*)alloc(128 * sizeof(float));
    float* shift  = (float*)alloc(128 * sizeof(float));
    bf16*  W1b    = (bf16*) alloc(H * H * sizeof(bf16));
    bf16*  Ws1b   = (bf16*) alloc(H * H * sizeof(bf16));
    bf16*  Wn1b   = (bf16*) alloc(H * H * sizeof(bf16));
    bf16*  W2b    = (bf16*) alloc(H * H * sizeof(bf16));
    bf16*  Ws2b   = (bf16*) alloc(H * H * sizeof(bf16));
    bf16*  Wn2b   = (bf16*) alloc(H * H * sizeof(bf16));
    bf16*  W3b    = (bf16*) alloc(64 * H * sizeof(bf16));

    const int ELT_BLK = 256;
    const int gemmBlocks = (N + 127) / 128;
    const long NH4 = NH / 4;
    const int eltBlocksNH4 = (int)((NH4 + ELT_BLK - 1) / ELT_BLK);

    // --- convert weights + input to bf16 -----------------------------------
    f2bf_kernel<<<eltBlocksNH4, ELT_BLK, 0, stream>>>(h, bfA, (int)NH4);
    f2bf_kernel<<<16, ELT_BLK, 0, stream>>>(W1,  W1b,  H * H / 4);
    f2bf_kernel<<<16, ELT_BLK, 0, stream>>>(Ws1, Ws1b, H * H / 4);
    f2bf_kernel<<<16, ELT_BLK, 0, stream>>>(Wn1, Wn1b, H * H / 4);
    f2bf_kernel<<<16, ELT_BLK, 0, stream>>>(W2,  W2b,  H * H / 4);
    f2bf_kernel<<<16, ELT_BLK, 0, stream>>>(Ws2, Ws2b, H * H / 4);
    f2bf_kernel<<<16, ELT_BLK, 0, stream>>>(Wn2, Wn2b, H * H / 4);
    f2bf_kernel<<<8,  ELT_BLK, 0, stream>>>(W3,  W3b,  64 * H / 4);

    // --- degrees (shared by both SAGE layers) ------------------------------
    hipMemsetAsync(degv, 0, (size_t)N * sizeof(float), stream);
    deg_kernel<<<(E + ELT_BLK - 1) / ELT_BLK, ELT_BLK, 0, stream>>>(dst, degv, E);

    // ===== layer 1: dense + BN + relu =====
    hipMemsetAsync(colsum, 0, 128 * sizeof(float), stream);
    hipMemsetAsync(colssq, 0, 128 * sizeof(float), stream);
    gemm_wmma<8, false, false, true, false><<<gemmBlocks, 256, 0, stream>>>(
        bfA, W1b, nullptr, nullptr, b1, big, nullptr, colsum, colssq, N);
    bn_finalize_kernel<<<1, 128, 0, stream>>>(colsum, colssq, g1, bt1, scale, shift, N);
    bn_apply_relu_kernel<<<eltBlocksNH4, ELT_BLK, 0, stream>>>(big, scale, shift, bfB, NH4);

    // ===== SAGE layer 1 =====
    hipMemsetAsync(big, 0, NH * sizeof(float), stream);
    agg_kernel<<<(E + 7) / 8, 256, 0, stream>>>(bfB, src, dst, big, E);
    mean_kernel<<<eltBlocksNH4, ELT_BLK, 0, stream>>>(big, degv, bfM, NH4);
    gemm_wmma<8, true, true, false, true><<<gemmBlocks, 256, 0, stream>>>(
        bfB, Ws1b, bfM, Wn1b, bs1, nullptr, bfA, nullptr, nullptr, N);

    // ===== layer 2: dense + BN + relu =====
    hipMemsetAsync(colsum, 0, 128 * sizeof(float), stream);
    hipMemsetAsync(colssq, 0, 128 * sizeof(float), stream);
    gemm_wmma<8, false, false, true, false><<<gemmBlocks, 256, 0, stream>>>(
        bfA, W2b, nullptr, nullptr, b2, big, nullptr, colsum, colssq, N);
    bn_finalize_kernel<<<1, 128, 0, stream>>>(colsum, colssq, g2, bt2, scale, shift, N);
    bn_apply_relu_kernel<<<eltBlocksNH4, ELT_BLK, 0, stream>>>(big, scale, shift, bfB, NH4);

    // ===== SAGE layer 2 =====
    hipMemsetAsync(big, 0, NH * sizeof(float), stream);
    agg_kernel<<<(E + 7) / 8, 256, 0, stream>>>(bfB, src, dst, big, E);
    mean_kernel<<<eltBlocksNH4, ELT_BLK, 0, stream>>>(big, degv, bfM, NH4);
    gemm_wmma<8, true, true, false, true><<<gemmBlocks, 256, 0, stream>>>(
        bfB, Ws2b, bfM, Wn2b, bs2, nullptr, bfA, nullptr, nullptr, N);

    // ===== output layer: relu(x @ W3^T + b3) -> [N, 64] f32 =====
    gemm_wmma<4, false, true, false, false><<<gemmBlocks, 256, 0, stream>>>(
        bfA, W3b, nullptr, nullptr, b3, (float*)d_out, nullptr, nullptr, nullptr, N);
}
